// Attention_19481971655316
// MI455X (gfx1250) — matrix operand order repaired and checked by reading, not compiled, not run on hardware
//
#include <hip/hip_runtime.h>
#include <stdint.h>

typedef __attribute__((ext_vector_type(16))) __bf16 v16bf;
typedef __attribute__((ext_vector_type(8)))  float  v8f;
typedef __attribute__((ext_vector_type(4)))  unsigned int u32x4;
typedef __attribute__((ext_vector_type(8)))  int i32x8;
typedef __attribute__((ext_vector_type(4)))  int i32x4;

#define WMMA_BF16(a, b, c) \
  __builtin_amdgcn_wmma_f32_16x16x32_bf16(false, (a), false, (b), (short)0, (c), false, false)

#if __has_builtin(__builtin_amdgcn_tensor_load_to_lds) && __has_builtin(__builtin_amdgcn_s_wait_tensorcnt)
#define HAVE_TDM 1
#else
#define HAVE_TDM 0
#endif

// ---------------------------------------------------------------------------
// WMMA fragment loaders (CDNA5 layouts, wave32)
// ---------------------------------------------------------------------------

// A fragment 16x32 bf16 from LDS laid out [row][k] (contiguous k).
// lane<16: K 0..15; lane>=16: K 16..31.
__device__ inline v16bf frag_a_rowk(const __bf16* lds, int stride, int row0, int k0, int lane) {
  int m   = lane & 15;
  int off = (lane >> 4) << 4;
  const __bf16* p = lds + (row0 + m) * stride + k0 + off;
  union { uint4 u[2]; v16bf v; } t;
  t.u[0] = *(const uint4*)(p);        // K = off .. off+7
  t.u[1] = *(const uint4*)(p + 8);    // K = off+8 .. off+15
  return t.v;
}

// A fragment 16x32 where the logical A is the transpose of LDS [k][m] layout
// (used for k^T in the context update). Scalar strided loads.
__device__ inline v16bf frag_a_colk(const __bf16* lds, int stride, int row0, int k0, int lane) {
  int m   = lane & 15;
  int off = (lane >> 4) << 4;
  v16bf a;
#pragma unroll
  for (int j = 0; j < 16; ++j) {
    int kk = off + j;
    a[j] = lds[(k0 + kk) * stride + row0 + m];
  }
  return a;
}

// B fragment 32x16 bf16 from LDS laid out [n][k] (contiguous k => vector loads).
// lane<16: K 0..15; lane>=16: K 16..31. Element j <-> K = kb + j.
__device__ inline v16bf frag_b_rowk(const __bf16* lds, int stride, int k0, int col0, int lane) {
  int n  = lane & 15;
  int kb = (lane >> 4) << 4;
  const __bf16* p = lds + (col0 + n) * stride + k0 + kb;
  union { uint4 u[2]; v16bf v; } t;
  t.u[0] = *(const uint4*)(p);
  t.u[1] = *(const uint4*)(p + 8);
  return t.v;
}

// B fragment 32x16 from LDS laid out [k][n] (strided). Used for v tile.
__device__ inline v16bf frag_b_colk(const __bf16* lds, int stride, int k0, int col0, int lane) {
  int n  = lane & 15;
  int kb = (lane >> 4) << 4;
  v16bf b;
#pragma unroll
  for (int j = 0; j < 16; ++j) b[j] = lds[(k0 + kb + j) * stride + col0 + n];
  return b;
}

#if HAVE_TDM
// ---------------------------------------------------------------------------
// TDM: DMA a 64-row x 128-byte tile from global into LDS with a 160-byte
// LDS row stride (pad 8 DWORDs after every 32 DWORDs stored).
// D# group0: count=1, lds_addr, 57-bit global addr, type=2.
// D# group1: data_size=8B; tensor_dim0=16 units, tensor_dim1=4096 rows,
//            tile_dim0=16 units, tile_dim1=64 rows, dim0_stride=16 units,
//            pad_enable, pad_interval=32 DW (code 4), pad_amount=8 DW (code 7).
// ---------------------------------------------------------------------------
__device__ inline void tdm_load_tile(unsigned lds_off, const void* gptr) {
  unsigned long long ga = (unsigned long long)gptr;
  u32x4 g0;
  g0[0] = 1u;                                                // count=1, user mode
  g0[1] = lds_off;                                           // lds_addr
  g0[2] = (unsigned)(ga & 0xffffffffu);                      // global_addr[31:0]
  g0[3] = (unsigned)((ga >> 32) & 0x1ffffffu) | (2u << 30);  // addr[56:32] | type=2
  i32x8 g1;
  g1[0] = (int)((3u << 16) | (1u << 20) | (4u << 22) | (7u << 25));
  g1[1] = (int)(16u << 16);      // tensor_dim0 = 16 (8B units)
  g1[2] = (int)(4096u << 16);    // tensor_dim0 hi=0 | tensor_dim1 lo = 4096
  g1[3] = (int)(16u << 16);      // tensor_dim1 hi=0 | tile_dim0 = 16
  g1[4] = 64;                    // tile_dim1 = 64, tile_dim2 = 0
  g1[5] = 16;                    // tensor_dim0_stride = 16 (8B units)
  g1[6] = 0;
  g1[7] = 0;
  i32x4 z4 = {0, 0, 0, 0};
  i32x8 z8 = {0, 0, 0, 0, 0, 0, 0, 0};
  __builtin_amdgcn_tensor_load_to_lds(g0, g1, z4, z4, z8, 0);
}
#endif

// ---------------------------------------------------------------------------
// f32 -> bf16 convert
// ---------------------------------------------------------------------------
__global__ void cvt_f32_bf16(const float* __restrict__ in, unsigned short* __restrict__ out, int n) {
  int i = blockIdx.x * 256 + threadIdx.x;
  if (i < n) ((__bf16*)out)[i] = (__bf16)in[i];
}

// ---------------------------------------------------------------------------
// QKV GEMM: Out[m,n] = sum_k A[m,k] * W[n,k].  M=16384, N=3072, K=1024.
// 64x64 tile per block, 8 waves x 2 C-tiles, BK=32 (one bf16 WMMA K-step).
// ---------------------------------------------------------------------------
__global__ __launch_bounds__(256) void gemm_qkv(const unsigned short* __restrict__ Ap,
                                                const unsigned short* __restrict__ Wp,
                                                unsigned short* __restrict__ Outp) {
  constexpr int K = 1024, N = 3072, STR = 48;
  const __bf16* A = (const __bf16*)Ap;
  const __bf16* W = (const __bf16*)Wp;
  __bf16* Out = (__bf16*)Outp;

  __shared__ __bf16 As[64 * STR];
  __shared__ __bf16 Bs[64 * STR];

  int t = threadIdx.x, lane = t & 31, wid = t >> 5;
  int m0 = blockIdx.y * 64, n0 = blockIdx.x * 64;
  int lrow = t >> 2, lk = (t & 3) * 8;
  int tr = wid & 3, tcp = (wid >> 2) * 2;

  v8f acc0 = {}, acc1 = {};

  for (int kt = 0; kt < K / 32; ++kt) {
    int kg = kt * 32 + lk;
    uint4 av = *(const uint4*)(A + (size_t)(m0 + lrow) * K + kg);
    uint4 bv = *(const uint4*)(W + (size_t)(n0 + lrow) * K + kg);
    __syncthreads();
    *(uint4*)&As[lrow * STR + lk] = av;
    *(uint4*)&Bs[lrow * STR + lk] = bv;
    __syncthreads();
    v16bf a  = frag_a_rowk(As, STR, tr * 16, 0, lane);
    v16bf b0 = frag_b_rowk(Bs, STR, 0, tcp * 16, lane);
    v16bf b1 = frag_b_rowk(Bs, STR, 0, (tcp + 1) * 16, lane);
    acc0 = WMMA_BF16(a, b0, acc0);
    acc1 = WMMA_BF16(a, b1, acc1);
  }

  int nn = lane & 15, mh = (lane >> 4) << 3;
#pragma unroll
  for (int r = 0; r < 8; ++r) {
    int m = m0 + tr * 16 + r + mh;
    Out[(size_t)m * N + n0 + tcp * 16 + nn]       = (__bf16)acc0[r];
    Out[(size_t)m * N + n0 + (tcp + 1) * 16 + nn] = (__bf16)acc1[r];
  }
}

// ---------------------------------------------------------------------------
// Feature maps: q = softmax(q)*E^-0.5, k = exp(k), v copy.
// One wave per (b,h,n) row of 64; lane handles 2 elements.
// qkv layout: [(b*N+n)*3072 + s*1024 + h*64 + e]; outputs (B,H,N,e).
// ---------------------------------------------------------------------------
__global__ __launch_bounds__(256) void feature_kernel(const unsigned short* __restrict__ QKVp,
                                                      unsigned short* __restrict__ Qfp,
                                                      unsigned short* __restrict__ Kfp,
                                                      unsigned short* __restrict__ Vfp) {
  const __bf16* QKV = (const __bf16*)QKVp;
  __bf16* Qf = (__bf16*)Qfp;
  __bf16* Kf = (__bf16*)Kfp;
  __bf16* Vf = (__bf16*)Vfp;

  int t = threadIdx.x, lane = t & 31, wid = t >> 5;
  long r = (long)blockIdx.x * 8 + wid;           // row over B*H*N = 262144
  int b = (int)(r >> 16);
  int rem = (int)(r & 65535);
  int h = rem >> 12;
  int n = rem & 4095;
  size_t sb = ((size_t)(b * 4096 + n)) * 3072 + h * 64;
  size_t db = (size_t)r * 64;

  // q: softmax over 64 values, scale by 1/sqrt(64)=0.125
  float q0 = (float)QKV[sb + 2 * lane], q1 = (float)QKV[sb + 2 * lane + 1];
  float mx = fmaxf(q0, q1);
#pragma unroll
  for (int o = 16; o > 0; o >>= 1) mx = fmaxf(mx, __shfl_xor(mx, o, 32));
  float e0 = __expf(q0 - mx), e1 = __expf(q1 - mx);
  float s = e0 + e1;
#pragma unroll
  for (int o = 16; o > 0; o >>= 1) s += __shfl_xor(s, o, 32);
  float inv = 0.125f / s;
  Qf[db + 2 * lane]     = (__bf16)(e0 * inv);
  Qf[db + 2 * lane + 1] = (__bf16)(e1 * inv);
  // k: exp
  float k0 = (float)QKV[sb + 1024 + 2 * lane], k1 = (float)QKV[sb + 1024 + 2 * lane + 1];
  Kf[db + 2 * lane]     = (__bf16)__expf(k0);
  Kf[db + 2 * lane + 1] = (__bf16)__expf(k1);
  // v: copy
  Vf[db + 2 * lane]     = QKV[sb + 2048 + 2 * lane];
  Vf[db + 2 * lane + 1] = QKV[sb + 2048 + 2 * lane + 1];
}

// ---------------------------------------------------------------------------
// Fused bucketed linear attention. One block per (b,h); sequential cumsum over
// U=64 buckets. context (64x64 f32) lives in LDS the whole time.
// q/k/v bucket tiles are DMA'd into LDS by the Tensor Data Mover (wave 0),
// with D# padding mapping 128B rows onto the 160B LDS stride.
// ---------------------------------------------------------------------------
__global__ __launch_bounds__(256) void attn_kernel(const unsigned short* __restrict__ Qfp,
                                                   const unsigned short* __restrict__ Kfp,
                                                   const unsigned short* __restrict__ Vfp,
                                                   unsigned short* __restrict__ Attnp) {
  constexpr int QS = 80;   // bf16 LDS stride (160B, 16B aligned)
  constexpr int CS = 68;   // f32 ctx stride
  const __bf16* Qf = (const __bf16*)Qfp;
  const __bf16* Kf = (const __bf16*)Kfp;
  const __bf16* Vf = (const __bf16*)Vfp;
  __bf16* Attn = (__bf16*)Attnp;

  __shared__ __bf16 qs[64 * QS];
  __shared__ __bf16 ks[64 * QS];
  __shared__ __bf16 vs[64 * QS];
  __shared__ __bf16 cbt[64 * QS];   // context transposed, bf16: cbt[e2][d]
  __shared__ float  ctx[64 * CS];   // context f32: ctx[d][e2]
  __shared__ float  ksum[64];
  __shared__ float  dinv[64];

  int bh = blockIdx.x;
  const size_t base = (size_t)bh * 4096 * 64;
  int t = threadIdx.x, lane = t & 31, wid = t >> 5;
  int tr = wid & 3, tcp = (wid >> 2) * 2;

  for (int i = t; i < 64 * CS; i += 256) ctx[i] = 0.0f;
  if (t < 64) ksum[t] = 0.0f;

  for (int u = 0; u < 64; ++u) {
    __syncthreads();  // prev iteration ctx/ksum updates + tile consumption done
#if HAVE_TDM
    if (wid == 0) {
      tdm_load_tile((unsigned)(size_t)(void*)qs, Qf + base + (size_t)u * 4096);
      tdm_load_tile((unsigned)(size_t)(void*)ks, Kf + base + (size_t)u * 4096);
      tdm_load_tile((unsigned)(size_t)(void*)vs, Vf + base + (size_t)u * 4096);
      __builtin_amdgcn_s_wait_tensorcnt(0);
    }
#else
    {
      const uint4* gq = (const uint4*)(Qf + base + (size_t)u * 4096);
      const uint4* gk = (const uint4*)(Kf + base + (size_t)u * 4096);
      const uint4* gv = (const uint4*)(Vf + base + (size_t)u * 4096);
      for (int i = t; i < 512; i += 256) {
        int row = i >> 3, c8 = i & 7;
        *(uint4*)&qs[row * QS + c8 * 8] = gq[i];
        *(uint4*)&ks[row * QS + c8 * 8] = gk[i];
        *(uint4*)&vs[row * QS + c8 * 8] = gv[i];
      }
    }
#endif
    // context f32 -> bf16, transposed so B-fragments are contiguous loads
    for (int i = t; i < 4096; i += 256) {
      int rr = i >> 6, cc = i & 63;
      cbt[cc * QS + rr] = (__bf16)ctx[rr * CS + cc];
    }
    if (u + 1 < 64) {
      __builtin_prefetch(Qf + base + (size_t)(u + 1) * 4096, 0, 1);
      __builtin_prefetch(Kf + base + (size_t)(u + 1) * 4096, 0, 1);
      __builtin_prefetch(Vf + base + (size_t)(u + 1) * 4096, 0, 1);
    }
    __syncthreads();
    // D_inv[n] = 1/max(ksum_prev . q[n], eps)
    if (t < 64) {
      float s = 0.0f;
      for (int d = 0; d < 64; ++d) s += ksum[d] * (float)qs[t * QS + d];
      dinv[t] = 1.0f / fmaxf(s, 1e-3f);
    }
    __syncthreads();
    // attn[n,e2] = D_inv[n] * sum_d q[n,d]*ctx_prev[d,e2]   (WMMA, 2 tiles/wave)
#pragma unroll
    for (int i = 0; i < 2; ++i) {
      int tc = tcp + i;
      v8f acc = {};
#pragma unroll
      for (int kt = 0; kt < 2; ++kt) {
        v16bf a = frag_a_rowk(qs, QS, tr * 16, kt * 32, lane);
        v16bf b = frag_b_rowk(cbt, QS, kt * 32, tc * 16, lane);  // cbt is [n][k]
        acc = WMMA_BF16(a, b, acc);
      }
      int nn = tc * 16 + (lane & 15);
      int mh = (lane >> 4) << 3;
#pragma unroll
      for (int r = 0; r < 8; ++r) {
        int m = tr * 16 + r + mh;
        float v = acc[r] * dinv[m];
        Attn[base + (size_t)(u * 64 + m) * 64 + nn] = (__bf16)v;
      }
    }
    // ctx[d,e2] += sum_n k[n,d]*v[n,e2]   (WMMA accumulate in LDS f32)
#pragma unroll
    for (int i = 0; i < 2; ++i) {
      int tc = tcp + i;
      int nn = tc * 16 + (lane & 15);
      int mh = (lane >> 4) << 3;
      v8f acc;
#pragma unroll
      for (int r = 0; r < 8; ++r) acc[r] = ctx[(tr * 16 + r + mh) * CS + nn];
#pragma unroll
      for (int kt = 0; kt < 2; ++kt) {
        v16bf a = frag_a_colk(ks, QS, tr * 16, kt * 32, lane);   // k^T
        v16bf b = frag_b_colk(vs, QS, kt * 32, tc * 16, lane);
        acc = WMMA_BF16(a, b, acc);
      }
#pragma unroll
      for (int r = 0; r < 8; ++r) ctx[(tr * 16 + r + mh) * CS + nn] = acc[r];
    }
    // ksum[d] += sum_n k[n,d]  (next-iter dinv consumes after the loop barrier)
    if (t < 64) {
      float s = 0.0f;
      for (int n2 = 0; n2 < 64; ++n2) s += (float)ks[n2 * QS + t];
      ksum[t] += s;
    }
  }
}

// ---------------------------------------------------------------------------
// Output projection: Out[m,n] = sum_k A[m,k]*W[n,k] + bias[n], f32 output.
// A is the attn buffer in (B,H,Nseq,e) layout; transpose folded into gather.
// M=16384, N=1024, K=1024.
// ---------------------------------------------------------------------------
__global__ __launch_bounds__(256) void gemm_proj(const unsigned short* __restrict__ Ap,
                                                 const unsigned short* __restrict__ Wp,
                                                 const float* __restrict__ bias,
                                                 float* __restrict__ Out) {
  constexpr int K = 1024, N = 1024, STR = 48;
  const __bf16* A = (const __bf16*)Ap;
  const __bf16* W = (const __bf16*)Wp;

  __shared__ __bf16 As[64 * STR];
  __shared__ __bf16 Bs[64 * STR];

  int t = threadIdx.x, lane = t & 31, wid = t >> 5;
  int m0 = blockIdx.y * 64, n0 = blockIdx.x * 64;
  int lrow = t >> 2, lk = (t & 3) * 8;
  int tr = wid & 3, tcp = (wid >> 2) * 2;

  v8f acc0 = {}, acc1 = {};

  for (int kt = 0; kt < K / 32; ++kt) {
    // A gather: logical (m = b*4096+ns, k = h*64+e) -> attn[(b*16+h)*4096+ns][e]
    int mg = m0 + lrow;
    int b = mg >> 12, ns = mg & 4095;
    int kg = kt * 32 + lk;
    int h = kg >> 6, e0 = kg & 63;
    uint4 av = *(const uint4*)(A + (((size_t)(b * 16 + h) * 4096 + ns) * 64 + e0));
    uint4 bv = *(const uint4*)(W + (size_t)(n0 + lrow) * K + kg);
    __syncthreads();
    *(uint4*)&As[lrow * STR + lk] = av;
    *(uint4*)&Bs[lrow * STR + lk] = bv;
    __syncthreads();
    v16bf a  = frag_a_rowk(As, STR, tr * 16, 0, lane);
    v16bf b0 = frag_b_rowk(Bs, STR, 0, tcp * 16, lane);
    v16bf b1 = frag_b_rowk(Bs, STR, 0, (tcp + 1) * 16, lane);
    acc0 = WMMA_BF16(a, b0, acc0);
    acc1 = WMMA_BF16(a, b1, acc1);
  }

  int nn = lane & 15, mh = (lane >> 4) << 3;
  float bi0 = bias[n0 + tcp * 16 + nn];
  float bi1 = bias[n0 + (tcp + 1) * 16 + nn];
#pragma unroll
  for (int r = 0; r < 8; ++r) {
    int m = m0 + tr * 16 + r + mh;
    Out[(size_t)m * N + n0 + tcp * 16 + nn]       = acc0[r] + bi0;
    Out[(size_t)m * N + n0 + (tcp + 1) * 16 + nn] = acc1[r] + bi1;
  }
}

// ---------------------------------------------------------------------------
// Launch
// ---------------------------------------------------------------------------
extern "C" void kernel_launch(void* const* d_in, const int* in_sizes, int n_in,
                              void* d_out, int out_size, void* d_ws, size_t ws_size,
                              hipStream_t stream) {
  (void)in_sizes; (void)n_in; (void)out_size; (void)ws_size;
  const float* x     = (const float*)d_in[0];  // (4,4096,1024)
  const float* qkvw  = (const float*)d_in[1];  // (3072,1024)
  const float* projw = (const float*)d_in[2];  // (1024,1024)
  const float* projb = (const float*)d_in[3];  // (1024,)
  float* out = (float*)d_out;                  // (4,4096,1024) f32

  char* ws = (char*)d_ws;
  size_t off = 0;
  auto carve = [&](size_t nbytes) -> unsigned short* {
    unsigned short* p = (unsigned short*)(ws + off);
    off = (off + nbytes + 255) & ~(size_t)255;
    return p;
  };
  unsigned short* x_bf   = carve(16777216ull * 2);  // x bf16
  unsigned short* wq_bf  = carve(3145728ull  * 2);  // qkv_w bf16
  unsigned short* wp_bf  = carve(1048576ull  * 2);  // proj_w bf16
  unsigned short* qkv_bf = carve(50331648ull * 2);  // qkv bf16 (B,N,3C)
  unsigned short* qf     = carve(16777216ull * 2);  // (B,H,N,e)
  unsigned short* kf     = carve(16777216ull * 2);
  unsigned short* vf     = carve(16777216ull * 2);
  unsigned short* attn   = carve(16777216ull * 2);

  cvt_f32_bf16<<<(16777216 + 255) / 256, 256, 0, stream>>>(x, x_bf, 16777216);
  cvt_f32_bf16<<<(3145728  + 255) / 256, 256, 0, stream>>>(qkvw, wq_bf, 3145728);
  cvt_f32_bf16<<<(1048576  + 255) / 256, 256, 0, stream>>>(projw, wp_bf, 1048576);

  gemm_qkv<<<dim3(48, 256), 256, 0, stream>>>(x_bf, wq_bf, qkv_bf);
  feature_kernel<<<32768, 256, 0, stream>>>(qkv_bf, qf, kf, vf);
  attn_kernel<<<64, 256, 0, stream>>>(qf, kf, vf, attn);
  gemm_proj<<<dim3(16, 256), 256, 0, stream>>>(attn, wp_bf, projb, out);
}
